// AFMExpert_88055419502698
// MI455X (gfx1250) — compile-verified, matrix operand order checked by reading, not run on hardware
//
#include <hip/hip_runtime.h>
#include <hip/hip_bf16.h>

#define BATCH    512
#define FDIM     64
#define EDIM     64
#define NPAIR    2016   // 64*63/2
#define NTILE    126    // NPAIR / 16
#define NWAVE    4
#define NTHREADS 128

typedef __attribute__((ext_vector_type(16))) __bf16 v16bf;
typedef __attribute__((ext_vector_type(8)))  float  v8f;

// Operand fragment for v_wmma_f32_16x16x32_bf16 (wave32). Same per-lane data
// serves as A (lane = row M) or B (lane = col N); K pattern per lane:
//   laneHalf==0 -> K = kbase + {0..7, 16..23} ; laneHalf==1 -> K = kbase + {8..15, 24..31}
__device__ __forceinline__ v16bf frag_from_bf16(const __bf16* row, int kbase, int laneHalf) {
  const int k0 = kbase + laneHalf * 8;
  v16bf r;
#pragma unroll
  for (int t = 0; t < 8; ++t) r[t] = row[k0 + t];
#pragma unroll
  for (int t = 0; t < 8; ++t) r[8 + t] = row[k0 + 16 + t];
  return r;
}

__global__ __launch_bounds__(NTHREADS)
void AFMExpert_88055419502698_kernel(const float* __restrict__ x,
                                     const float* __restrict__ w1, const float* __restrict__ b1,
                                     const float* __restrict__ w2, const float* __restrict__ b2,
                                     const float* __restrict__ pw, const float* __restrict__ pb,
                                     float* __restrict__ out) {
  __shared__ float          xs[FDIM * EDIM];       // x[b], f32: 16 KB
  __shared__ __bf16         w1s[64 * 64];          // 8 KB
  __shared__ __bf16         w2s[64 * 64];          // 8 KB
  __shared__ float          b1s[64], b2s[64];
  __shared__ unsigned short pairIJ[NPAIR];         // (i<<8)|j per pair: 4 KB
  __shared__ float          redS[NWAVE * 64 * 16]; // epilogue reduction transpose: 16 KB
  __shared__ float          dsum[64];
  __shared__ float          afm[64];

  const int b        = blockIdx.x;
  const int tid      = threadIdx.x;
  const int wave     = tid >> 5;
  const int lane     = tid & 31;
  const int laneRow  = lane & 15;   // pair index within tile (matmuls run transposed)
  const int laneHalf = lane >> 4;
  const int k0       = 8 * laneHalf;

  // ---- stage x[b], weights (bf16), biases, pair table ----
  const float* xb = x + (size_t)b * FDIM * EDIM;
  for (int idx = tid; idx < FDIM * EDIM; idx += NTHREADS) {
    xs[idx]  = xb[idx];
    w1s[idx] = (__bf16)w1[idx];
    w2s[idx] = (__bf16)w2[idx];
  }
  if (tid < 64) { b1s[tid] = b1[tid]; b2s[tid] = b2[tid]; }
  for (int p = tid; p < NPAIR; p += NTHREADS) {    // unrank triu(64,1) once
    int i = 0, pr = p;
    while (pr >= 63 - i) { pr -= 63 - i; ++i; }
    pairIJ[p] = (unsigned short)((i << 8) | (i + 1 + pr));
  }
  __syncthreads();

  // ---- preload weight A-fragments (loop-invariant) ----
  v16bf w1f[2][4], w2f[2][4];
#pragma unroll
  for (int hb = 0; hb < 4; ++hb) {
    const int r = hb * 16 + laneRow;
#pragma unroll
    for (int kc = 0; kc < 2; ++kc) {
      w1f[kc][hb] = frag_from_bf16(&w1s[r * 64], kc * 32, laneHalf);
      w2f[kc][hb] = frag_from_bf16(&w2s[r * 64], kc * 32, laneHalf);
    }
  }
  // ---- preload biases per output row (m = hb*16 + v + 8*laneHalf) ----
  float b1v[4][8], b2v[4][8];
#pragma unroll
  for (int hb = 0; hb < 4; ++hb)
#pragma unroll
    for (int v = 0; v < 8; ++v) {
      b1v[hb][v] = b1s[hb * 16 + v + k0];
      b2v[hb][v] = b2s[hb * 16 + v + k0];
    }

  // Per-lane exp-sum accumulators over this lane's pairs (p = t*16 + laneRow):
  // index o = ob*16 + v + 8*laneHalf. Softmax needs no max-shift: logits are O(1),
  // f32 exp is safe to ~88, and softmax is shift-invariant.
  float dacc[4][8], sacc[4][8];
#pragma unroll
  for (int ob = 0; ob < 4; ++ob)
#pragma unroll
    for (int v = 0; v < 8; ++v) { dacc[ob][v] = 0.f; sacc[ob][v] = 0.f; }

  for (int t = wave; t < NTILE; t += NWAVE) {
    // ---- hadamard values computed DIRECTLY in fragment order: no LDS round trip.
    // hv[ob][t8] = had[p=laneRow, e = ob*16 + k0 + t8] = xi[e]*xj[e]
    const unsigned int ij = pairIJ[t * 16 + laneRow];
    const float* xi = &xs[(ij >> 8) * 64];
    const float* xj = &xs[(ij & 255u) * 64];
    float hv[4][8];
#pragma unroll
    for (int ob = 0; ob < 4; ++ob)
#pragma unroll
      for (int t8 = 0; t8 < 8; ++t8) {
        const int q = ob * 16 + k0 + t8;
        hv[ob][t8] = xi[q] * xj[q];
      }
    v16bf ha0, ha1;                                 // B operand: hadT (lane = col p)
#pragma unroll
    for (int t8 = 0; t8 < 8; ++t8) {
      ha0[t8]     = (__bf16)hv[0][t8];
      ha0[8 + t8] = (__bf16)hv[1][t8];
      ha1[t8]     = (__bf16)hv[2][t8];
      ha1[8 + t8] = (__bf16)hv[3][t8];
    }

    // ---- layer 1 (transposed): aT = relu(W1 @ hadT + b1), stays in registers ----
    float c1[4][8];
#pragma unroll
    for (int hb = 0; hb < 4; ++hb) {
      v8f c;
#pragma unroll
      for (int v = 0; v < 8; ++v) c[v] = b1v[hb][v];
      c = __builtin_amdgcn_wmma_f32_16x16x32_bf16(false, w1f[0][hb], false, ha0, (short)0, c, false, false);
      c = __builtin_amdgcn_wmma_f32_16x16x32_bf16(false, w1f[1][hb], false, ha1, (short)0, c, false, false);
#pragma unroll
      for (int v = 0; v < 8; ++v) c1[hb][v] = c[v] > 0.f ? c[v] : 0.f;
    }
    // repack aT (C layout) directly into layer-2 B fragments
    v16bf aa0, aa1;
#pragma unroll
    for (int t8 = 0; t8 < 8; ++t8) {
      aa0[t8]     = (__bf16)c1[0][t8];
      aa0[8 + t8] = (__bf16)c1[1][t8];
      aa1[t8]     = (__bf16)c1[2][t8];
      aa1[8 + t8] = (__bf16)c1[3][t8];
    }

    // ---- layer 2 (transposed) + exp accumulation ----
#pragma unroll
    for (int ob = 0; ob < 4; ++ob) {
      v8f c;
#pragma unroll
      for (int v = 0; v < 8; ++v) c[v] = b2v[ob][v];
      c = __builtin_amdgcn_wmma_f32_16x16x32_bf16(false, w2f[0][ob], false, aa0, (short)0, c, false, false);
      c = __builtin_amdgcn_wmma_f32_16x16x32_bf16(false, w2f[1][ob], false, aa1, (short)0, c, false, false);
#pragma unroll
      for (int v = 0; v < 8; ++v) {
        const float e = __expf(c[v]);               // logits[o, p=laneRow]
        dacc[ob][v] += e;
        sacc[ob][v] += e * hv[ob][v];
      }
    }
  }

  // ---- reduce over pair lanes: deterministic transpose through LDS ----
  __syncthreads();
  // pass 1: denominators
#pragma unroll
  for (int ob = 0; ob < 4; ++ob)
#pragma unroll
    for (int v = 0; v < 8; ++v)
      redS[wave * 1024 + (ob * 16 + v + k0) * 16 + laneRow] = dacc[ob][v];
  __syncthreads();
  if (tid < 64) {
    float D = 0.f;
    for (int w = 0; w < NWAVE; ++w)
      for (int i = 0; i < 16; ++i) D += redS[w * 1024 + tid * 16 + i];
    dsum[tid] = D;
  }
  __syncthreads();
  // pass 2: numerators
#pragma unroll
  for (int ob = 0; ob < 4; ++ob)
#pragma unroll
    for (int v = 0; v < 8; ++v)
      redS[wave * 1024 + (ob * 16 + v + k0) * 16 + laneRow] = sacc[ob][v];
  __syncthreads();
  if (tid < 64) {
    float S = 0.f;
    for (int w = 0; w < NWAVE; ++w)
      for (int i = 0; i < 16; ++i) S += redS[w * 1024 + tid * 16 + i];
    afm[tid] = S / dsum[tid];                       // afm[b, tid]
  }
  __syncthreads();

  if (tid == 0) {
    float acc = pb[0];
    for (int e = 0; e < 64; ++e) acc += afm[e] * pw[e];
    out[b] = acc;                                   // out shape [B, 1]
  }
}

extern "C" void kernel_launch(void* const* d_in, const int* in_sizes, int n_in,
                              void* d_out, int out_size, void* d_ws, size_t ws_size,
                              hipStream_t stream) {
  const float* x  = (const float*)d_in[0];
  const float* w1 = (const float*)d_in[1];   // attn_w_w [NHID,E]
  const float* b1 = (const float*)d_in[2];   // attn_w_b [NHID]
  const float* w2 = (const float*)d_in[3];   // attn_h_w [OUT,NHID]
  const float* b2 = (const float*)d_in[4];   // attn_h_b [OUT]
  const float* pw = (const float*)d_in[5];   // attn_p_w [1,E]
  const float* pb = (const float*)d_in[6];   // attn_p_b [1]
  float* out = (float*)d_out;                // [B,1]
  hipLaunchKernelGGL(AFMExpert_88055419502698_kernel, dim3(BATCH), dim3(NTHREADS), 0, stream,
                     x, w1, b1, w2, b2, pw, pb, out);
}
